// NODE_80307298501196
// MI455X (gfx1250) — matrix operand order repaired and checked by reading, not compiled, not run on hardware
//
#include <hip/hip_runtime.h>

// MI455X (gfx1250) implementation.
// Roofline: output GEMM (256x524288 @ 524288x100, 26.8 GFLOP) is bandwidth
// bound: out_W 210MB + combined 2x268MB(f16) ~ 0.75GB -> ~30us @ 23.3TB/s.
// -> f16 WMMA (v_wmma_f32_16x16x32_f16), f32 accumulate, split-K + f32 atomics.
// Workspace layout (needs ~304 MiB):
//   [0, 48MiB)         fv   : fp32, (4 layers, 256, 12288) sigmoid probabilities
//   [48MiB, 304MiB)    comb : f16,  (256, 524288) leaf features, row major

typedef __attribute__((ext_vector_type(16))) _Float16 v16h;
typedef __attribute__((ext_vector_type(8)))  _Float16 v8h;
typedef __attribute__((ext_vector_type(8)))  float    v8f;

#define B_   256
#define F_   128
#define NL   4
#define T_   2048
#define DEP  6
#define L_   64
#define C_   100
#define ND   (T_ * DEP)        // 12288 selector outputs per layer
#define KTOT (NL * T_ * L_)    // 524288 combined features
#define LSTR (T_ * L_)         // 131072 per-layer stride in combined
#define KC   2048              // K chunk per wave in output GEMM

__device__ __forceinline__ v16h pack16(float4 g0, float4 g1, float4 g2, float4 g3) {
  v16h r;
  r[0]  = (_Float16)g0.x; r[1]  = (_Float16)g0.y; r[2]  = (_Float16)g0.z; r[3]  = (_Float16)g0.w;
  r[4]  = (_Float16)g1.x; r[5]  = (_Float16)g1.y; r[6]  = (_Float16)g1.z; r[7]  = (_Float16)g1.w;
  r[8]  = (_Float16)g2.x; r[9]  = (_Float16)g2.y; r[10] = (_Float16)g2.z; r[11] = (_Float16)g2.w;
  r[12] = (_Float16)g3.x; r[13] = (_Float16)g3.y; r[14] = (_Float16)g3.z; r[15] = (_Float16)g3.w;
  return r;
}

// ---------------------------------------------------------------------------
// Kernel A: fv = sigmoid(x @ sel_W[i]^T + sel_b[i]); one wave = one 16x16 tile
// A row (x) and B row (sel_W[layer][n]) are both contiguous in K=F.
// ---------------------------------------------------------------------------
__global__ __launch_bounds__(128) void selector_gemm(
    const float* __restrict__ x, const float* __restrict__ selW,
    const float* __restrict__ selb, float* __restrict__ fv) {
  const int lane = threadIdx.x & 31;
  const int wave = threadIdx.x >> 5;
  const int tile = blockIdx.x * 4 + wave;           // 49152 tiles total
  const int tiles_per_layer = 16 * (ND / 16);       // 12288
  const int layer = tile / tiles_per_layer;
  const int r     = tile % tiles_per_layer;
  const int mt    = r / (ND / 16);
  const int nt    = r % (ND / 16);

  const int half = lane >> 4;
  const int m    = mt * 16 + (lane & 15);           // A row for this lane
  const int n    = nt * 16 + (lane & 15);           // B col for this lane

  const float* arow = x + (size_t)m * F_;
  const float* brow = selW + (size_t)layer * ND * F_ + (size_t)n * F_;

  v8f c = {};
  #pragma unroll
  for (int k = 0; k < F_; k += 32) {
    // A fragment: lane half h holds K = {h*8..h*8+7} U {16+h*8..16+h*8+7}
    const float4* pa0 = (const float4*)(arow + k + half * 8);
    const float4* pa1 = (const float4*)(arow + k + 16 + half * 8);
    v16h a = pack16(pa0[0], pa0[1], pa1[0], pa1[1]);
    // B fragment: lane half h holds K = {h*8..h*8+7} U {16+h*8..16+h*8+7}
    const float4* pb0 = (const float4*)(brow + k + half * 8);
    const float4* pb1 = (const float4*)(brow + k + 16 + half * 8);
    v16h b = pack16(pb0[0], pb0[1], pb1[0], pb1[1]);
    c = __builtin_amdgcn_wmma_f32_16x16x32_f16(false, a, false, b, (short)0, c,
                                               false, false);
  }

  const float bias = selb[layer * ND + n];
  const int rowbase = layer * B_ + mt * 16 + half * 8;
  const int colg    = nt * 16 + (lane & 15);
  #pragma unroll
  for (int v = 0; v < 8; ++v) {
    float z = c[v] + bias;
    float s = 1.0f / (1.0f + __expf(-z));
    fv[(size_t)(rowbase + v) * ND + colg] = s;
  }
}

// ---------------------------------------------------------------------------
// Kernel B: leaf products. One thread = 8 leaves of one (layer,b,tree).
// bit_d(l) = (l >> (5-d)) & 1 ; factor = bit ? (1-p_d) : p_d
// prefix = depths 0..2 (from leaf-group), tail = depths 3..5 (8 combos).
// ---------------------------------------------------------------------------
__global__ __launch_bounds__(256) void leaf_kernel(
    const float* __restrict__ fv, _Float16* __restrict__ comb) {
  const unsigned g = blockIdx.x * 256u + threadIdx.x;   // 16,777,216 threads
  const int lg    = g & 7;
  const int t     = (g >> 3) & (T_ - 1);
  const int layer = (g >> 14) & 3;
  const int b     = g >> 16;

  const float* p = fv + ((size_t)(layer * B_ + b) * ND + t * DEP);
  float2 p01 = *(const float2*)(p);
  float2 p23 = *(const float2*)(p + 2);
  float2 p45 = *(const float2*)(p + 4);
  const float p0 = p01.x, p1 = p01.y, p2 = p23.x;
  const float p3 = p23.y, p4 = p45.x, p5 = p45.y;

  const float f0 = (lg & 4) ? (1.0f - p0) : p0;
  const float f1 = (lg & 2) ? (1.0f - p1) : p1;
  const float f2 = (lg & 1) ? (1.0f - p2) : p2;
  const float pre = f0 * f1 * f2;

  const float q3 = 1.0f - p3, q4 = 1.0f - p4, q5 = 1.0f - p5;
  const float a0 = pre * p3 * p4, a1 = pre * p3 * q4;
  const float a2 = pre * q3 * p4, a3 = pre * q3 * q4;

  v8h o;
  o[0] = (_Float16)(a0 * p5); o[1] = (_Float16)(a0 * q5);
  o[2] = (_Float16)(a1 * p5); o[3] = (_Float16)(a1 * q5);
  o[4] = (_Float16)(a2 * p5); o[5] = (_Float16)(a2 * q5);
  o[6] = (_Float16)(a3 * p5); o[7] = (_Float16)(a3 * q5);

  *(v8h*)(comb + (size_t)b * KTOT + (size_t)layer * LSTR + t * L_ + lg * 8) = o;
}

// ---------------------------------------------------------------------------
// Kernel C0: out[b,c] = out_b[c]
// ---------------------------------------------------------------------------
__global__ void init_out(const float* __restrict__ outb, float* __restrict__ out) {
  int i = blockIdx.x * 256 + threadIdx.x;
  if (i < B_ * C_) out[i] = outb[i % C_];
}

// ---------------------------------------------------------------------------
// Kernel C: out += comb @ out_W^T, split-K with f32 atomics.
// Block = 8 waves = 8 M-tiles (same K chunk -> out_W L2 reuse).
// Each wave keeps all 7 N-tiles in registers -> comb read exactly once.
// ---------------------------------------------------------------------------
__global__ __launch_bounds__(256) void out_gemm(
    const _Float16* __restrict__ comb, const float* __restrict__ outW,
    float* __restrict__ out) {
  const int lane   = threadIdx.x & 31;
  const int wave   = threadIdx.x >> 5;
  const int mgroup = blockIdx.x & 1;
  const int kc     = blockIdx.x >> 1;            // 256 K-chunks of KC=2048
  const int mt     = mgroup * 8 + wave;
  const int half   = lane >> 4;
  const int m      = mt * 16 + (lane & 15);

  const _Float16* arow = comb + (size_t)m * KTOT;
  const int k0 = kc * KC;

  v8f acc[7];
  #pragma unroll
  for (int n = 0; n < 7; ++n) acc[n] = (v8f){};

  const float* brow[7];
  #pragma unroll
  for (int n = 0; n < 7; ++n) {
    int cc = n * 16 + (lane & 15);
    if (cc > C_ - 1) cc = C_ - 1;                // clamp: stay in-bounds, masked at store
    brow[n] = outW + (size_t)cc * KTOT + half * 16;
  }

  for (int k = k0; k < k0 + KC; k += 32) {
    v8h lo = *(const v8h*)(arow + k + half * 16);
    v8h hi = *(const v8h*)(arow + k + half * 16 + 8);
    v16h a;
    #pragma unroll
    for (int i = 0; i < 8; ++i) { a[i] = lo[i]; a[8 + i] = hi[i]; }
    #pragma unroll
    for (int n = 0; n < 7; ++n) {
      const float4* q = (const float4*)(brow[n] + k);
      v16h b = pack16(q[0], q[1], q[2], q[3]);
      acc[n] = __builtin_amdgcn_wmma_f32_16x16x32_f16(false, a, false, b,
                                                      (short)0, acc[n], false, false);
    }
  }

  const int colbase = lane & 15;
  #pragma unroll
  for (int n = 0; n < 7; ++n) {
    const int col = n * 16 + colbase;
    if (col < C_) {
      #pragma unroll
      for (int v = 0; v < 8; ++v) {
        const int row = mt * 16 + half * 8 + v;
        unsafeAtomicAdd(&out[row * C_ + col], acc[n][v]);
      }
    }
  }
}

// ---------------------------------------------------------------------------
extern "C" void kernel_launch(void* const* d_in, const int* in_sizes, int n_in,
                              void* d_out, int out_size, void* d_ws, size_t ws_size,
                              hipStream_t stream) {
  const float* x    = (const float*)d_in[0];
  const float* selW = (const float*)d_in[1];
  const float* selb = (const float*)d_in[2];
  const float* outW = (const float*)d_in[3];
  const float* outb = (const float*)d_in[4];
  float* out = (float*)d_out;

  float*     fv   = (float*)d_ws;                                  // 48 MiB
  _Float16*  comb = (_Float16*)((char*)d_ws +
                                (size_t)NL * B_ * ND * sizeof(float)); // 256 MiB

  // A: 4 layers * 16 m-tiles * 768 n-tiles = 49152 wave-tiles, 4 waves/block
  selector_gemm<<<12288, 128, 0, stream>>>(x, selW, selb, fv);
  // B: 256*4*2048*8 threads
  leaf_kernel<<<65536, 256, 0, stream>>>(fv, comb);
  // C
  init_out<<<(B_ * C_ + 255) / 256, 256, 0, stream>>>(outb, out);
  out_gemm<<<512, 256, 0, stream>>>(comb, outW, out);
}
